// MHLA_Video_Uni_40845138985075
// MI455X (gfx1250) — matrix operand order repaired and checked by reading, not compiled, not run on hardware
//
#include <hip/hip_runtime.h>
#include <hip/hip_bf16.h>

// ---------------- problem constants ----------------
#define F_T 12
#define H_T 30
#define W_T 50
#define FB 3
#define HB 5
#define WB 10
#define P1C 4
#define P2C 6
#define P3C 5
#define NBLK 150      // FB*HB*WB
#define NBLKP 160     // NBLK padded to multiple of 32 (WMMA K)
#define PTOK 120      // P1*P2*P3
#define NTOK 18000    // F_*H_*W_
#define MPAD 18048    // NTOK padded to multiple of 64 (GEMM M-tile)
#define DIM 1536
#define NH 12
#define HD 128
#define CC 64         // HD/2 rope pairs
#define C0S 22        // SPLITS[0]
#define C1S 43        // SPLITS[0]+SPLITS[1]
#define EPSF 1e-6f

typedef __bf16 bf16_t;
typedef __attribute__((ext_vector_type(16))) __bf16 v16bf;
typedef __attribute__((ext_vector_type(8)))  __bf16 v8bf;
typedef __attribute__((ext_vector_type(8)))  float  v8f;

__device__ __forceinline__ v8f wmma_bf16(v16bf a, v16bf b, v8f c) {
    return __builtin_amdgcn_wmma_f32_16x16x32_bf16(false, a, false, b, (short)0, c, false, false);
}
__device__ __forceinline__ v16bf cat8(v8bf lo, v8bf hi) {
    return __builtin_shufflevector(lo, hi, 0,1,2,3,4,5,6,7,8,9,10,11,12,13,14,15);
}

// ---------------- utility kernels ----------------
__global__ void zero_u32(unsigned int* __restrict__ p, size_t n) {
    size_t i = (size_t)blockIdx.x * 256 + threadIdx.x;
    if (i < n) p[i] = 0u;
}

__global__ void cvt_f32_bf16(const float* __restrict__ src, bf16_t* __restrict__ dst, size_t n) {
    size_t i = (size_t)blockIdx.x * 256 + threadIdx.x;
    if (i < n) dst[i] = (bf16_t)src[i];
}

// W_BLK[o][i] = (1 - d(o,i)/dmax) / colsum(i); dmax = dist of opposite grid corners
__global__ __launch_bounds__(256) void wblk_kernel(float* __restrict__ wblk) {
    const int i = blockIdx.x;          // column
    const int o = threadIdx.x;         // row candidate
    __shared__ float red[256];
    __shared__ float colv[NBLK];
    float m = 0.f;
    if (o < NBLK) {
        int fi = i / (HB*WB), ri = i % (HB*WB), hi = ri / WB, wi = ri % WB;
        int fo = o / (HB*WB), ro = o % (HB*WB), ho = ro / WB, wo = ro % WB;
        float df = (float)(fo - fi), dh = (float)(ho - hi), dw = (float)(wo - wi);
        float d = sqrtf(df*df + dh*dh + dw*dw);
        float dmax = sqrtf((float)((FB-1)*(FB-1) + (HB-1)*(HB-1) + (WB-1)*(WB-1)));
        m = 1.0f - d / dmax;
        colv[o] = m;
    }
    red[threadIdx.x] = m;
    __syncthreads();
    for (int s = 128; s > 0; s >>= 1) {
        if (threadIdx.x < s) red[threadIdx.x] += red[threadIdx.x + s];
        __syncthreads();
    }
    if (o < NBLK) wblk[o * NBLK + i] = colv[o] / red[0];
}

// bf16 W_BLK padded to 160x160 with zeros (rows=o, cols=i; K-contiguous)
__global__ void wblk_to_bf16(const float* __restrict__ wblk, bf16_t* __restrict__ wblkb) {
    int idx = blockIdx.x * 256 + threadIdx.x;
    if (idx >= NBLKP * NBLKP) return;
    const int i = idx % NBLKP;
    const int o = idx / NBLKP;
    wblkb[idx] = (o < NBLK && i < NBLK) ? (bf16_t)wblk[o * NBLK + i] : (bf16_t)0.f;
}

// ---------------- WMMA GEMM: C[M,Nout] = A[Mpad,K](bf16) * W[Nout,K]^T (bf16) + bias ----------------
// grid.x = Nout/64, grid.y = ceil(Mpad/256), block = 128 (4 waves), wave: 64 rows x 64 cols
__global__ __launch_bounds__(128) void gemm_bf16(const bf16_t* __restrict__ A,
                                                 const bf16_t* __restrict__ W,
                                                 const float* __restrict__ bias,
                                                 float* __restrict__ C,
                                                 int Mpad, int Mstore, int K, int Nout) {
    const int lane = threadIdx.x & 31;
    const int wave = threadIdx.x >> 5;
    const int m0 = (blockIdx.y * 4 + wave) * 64;
    if (m0 >= Mpad) return;                    // wave-uniform: EXEC stays full
    const int n0 = blockIdx.x * 64;
    const int col = lane & 15;
    const int half = lane >> 4;
    v8f acc[4][4] = {};
    const bf16_t* arow0 = A + (size_t)(m0 + col) * K;
    for (int k0 = 0; k0 < K; k0 += 32) {
        v16bf a[4];
#pragma unroll
        for (int mt = 0; mt < 4; ++mt) {
            const bf16_t* ar = arow0 + (size_t)mt * 16 * K;
            a[mt] = cat8(*(const v8bf*)(ar + k0 + 8*half),
                         *(const v8bf*)(ar + k0 + 16 + 8*half));
        }
        __builtin_prefetch(arow0 + k0 + 128, 0, 0);   // stream-ahead on A
#pragma unroll
        for (int nt = 0; nt < 4; ++nt) {
            const bf16_t* br = W + (size_t)(n0 + nt*16 + col) * K + k0;
            v16bf b = cat8(*(const v8bf*)(br + 8*half), *(const v8bf*)(br + 16 + 8*half));
#pragma unroll
            for (int mt = 0; mt < 4; ++mt)
                acc[mt][nt] = wmma_bf16(a[mt], b, acc[mt][nt]);
        }
    }
#pragma unroll
    for (int nt = 0; nt < 4; ++nt) {
        const int n = n0 + nt*16 + col;
        const float bv = bias[n];
#pragma unroll
        for (int mt = 0; mt < 4; ++mt)
#pragma unroll
            for (int r = 0; r < 8; ++r) {
                const int m = m0 + mt*16 + r + 8*half;
                if (m < Mstore) C[(size_t)m * Nout + n] = acc[mt][nt][r] + bv;
            }
    }
}

// ---------------- fused rmsnorm/relu/eps + rope + block scatter + ksum ----------------
// one block per token (256 threads)
__global__ __launch_bounds__(256) void post_kernel(float* __restrict__ q,
                                                   const float* __restrict__ k,
                                                   const float* __restrict__ v,
                                                   const float* __restrict__ nqw,
                                                   const float* __restrict__ nkw,
                                                   const float* __restrict__ fcos,
                                                   const float* __restrict__ fsin,
                                                   bf16_t* __restrict__ qrb,   // [h][nb][128p][128d]
                                                   bf16_t* __restrict__ krbT,  // [h][nb][128d][128p]
                                                   bf16_t* __restrict__ vbT,   // [h][nb][128e][128p]
                                                   float* __restrict__ ksum) { // [h][nb][128d]
    const int n = blockIdx.x;
    const int tid = threadIdx.x;
    const int f = n / (H_T * W_T);
    const int rem = n % (H_T * W_T);
    const int h = rem / W_T;
    const int w = rem % W_T;
    const int fb = f / P1C, p1 = f % P1C;
    const int hb = h / P2C, p2 = h % P2C;
    const int wb = w / P3C, p3 = w % P3C;
    const int nb = (fb * HB + hb) * WB + wb;
    const int p  = (p1 * P2C + p2) * P3C + p3;

    const float* qrow = q + (size_t)n * DIM;
    const float* krow = k + (size_t)n * DIM;
    const float* vrow = v + (size_t)n * DIM;

    __shared__ float red[256];
    float sq = 0.f, sk = 0.f;
    for (int c = tid; c < DIM; c += 256) {
        float a = qrow[c]; sq += a * a;
        float b = krow[c]; sk += b * b;
    }
    red[tid] = sq; __syncthreads();
    for (int s = 128; s > 0; s >>= 1) { if (tid < s) red[tid] += red[tid + s]; __syncthreads(); }
    const float rq = rsqrtf(red[0] / (float)DIM + EPSF);
    __syncthreads();
    red[tid] = sk; __syncthreads();
    for (int s = 128; s > 0; s >>= 1) { if (tid < s) red[tid] += red[tid + s]; __syncthreads(); }
    const float rk = rsqrtf(red[0] / (float)DIM + EPSF);
    __syncthreads();

    for (int pr = tid; pr < DIM / 2; pr += 256) {
        const int hh = pr / CC;
        const int ci = pr % CC;
        const int ch = hh * HD + 2 * ci;
        float q0 = qrow[ch]     * rq * nqw[ch];     q0 = (q0 > 0.f ? q0 : 0.f) + EPSF;
        float q1 = qrow[ch + 1] * rq * nqw[ch + 1]; q1 = (q1 > 0.f ? q1 : 0.f) + EPSF;
        float k0 = krow[ch]     * rk * nkw[ch];     k0 = (k0 > 0.f ? k0 : 0.f) + EPSF;
        float k1 = krow[ch + 1] * rk * nkw[ch + 1]; k1 = (k1 > 0.f ? k1 : 0.f) + EPSF;
        q[(size_t)n * DIM + ch]     = q0;     // normed q for qk kernel
        q[(size_t)n * DIM + ch + 1] = q1;
        const size_t hb_blk = (size_t)hh * NBLK + nb;
        atomicAdd(&ksum[hb_blk * HD + 2 * ci],     k0);
        atomicAdd(&ksum[hb_blk * HD + 2 * ci + 1], k1);
        const int row = (ci < C0S) ? f : ((ci < C1S) ? h : w);
        const float cs = fcos[row * CC + ci];
        const float sn = fsin[row * CC + ci];
        const float qr0 = q0 * cs - q1 * sn, qr1 = q0 * sn + q1 * cs;
        const float kr0 = k0 * cs - k1 * sn, kr1 = k0 * sn + k1 * cs;
        bf16_t* qd = qrb + (hb_blk * 128 + p) * 128;
        qd[2 * ci]     = (bf16_t)qr0;
        qd[2 * ci + 1] = (bf16_t)qr1;
        krbT[(hb_blk * 128 + 2 * ci) * 128 + p]     = (bf16_t)kr0;
        krbT[(hb_blk * 128 + 2 * ci + 1) * 128 + p] = (bf16_t)kr1;
    }
    for (int c = tid; c < DIM; c += 256) {
        const int hh = c / HD;
        const int e = c % HD;
        vbT[(((size_t)hh * NBLK + nb) * 128 + e) * 128 + p] = (bf16_t)vrow[c];
    }
}

// ---------------- kv: per (h,nb): kv[d,e] = sum_p krb[p,d]*vb[p,e]; stored K-major as kvT[h][de][i=nb] ----------------
__global__ __launch_bounds__(256) void kv_kernel(const bf16_t* __restrict__ krbT,
                                                 const bf16_t* __restrict__ vbT,
                                                 bf16_t* __restrict__ kvT) { // [h][16384 de][160 i]
    const size_t blk = blockIdx.x;
    const int hh = (int)(blk / NBLK);
    const int nb = (int)(blk % NBLK);
    const int lane = threadIdx.x & 31;
    const int wave = threadIdx.x >> 5;
    const int col = lane & 15;
    const int half = lane >> 4;
    const bf16_t* Ab = krbT + blk * 128 * 128;   // [d][p]
    const bf16_t* Bb = vbT + blk * 128 * 128;    // [e][p]
    const int m0 = wave * 16;
    v8f acc[8] = {};
    const bf16_t* arow = Ab + (size_t)(m0 + col) * 128;
#pragma unroll
    for (int k0 = 0; k0 < 128; k0 += 32) {
        v16bf a = cat8(*(const v8bf*)(arow + k0 + 8*half),
                       *(const v8bf*)(arow + k0 + 16 + 8*half));
#pragma unroll
        for (int t = 0; t < 8; ++t) {
            const bf16_t* br = Bb + (size_t)(t*16 + col) * 128 + k0;
            v16bf b = cat8(*(const v8bf*)(br + 8*half), *(const v8bf*)(br + 16 + 8*half));
            acc[t] = wmma_bf16(a, b, acc[t]);
        }
    }
#pragma unroll
    for (int t = 0; t < 8; ++t) {
        const int e = t*16 + col;
#pragma unroll
        for (int r = 0; r < 8; ++r) {
            const int d = m0 + r + 8*half;
            kvT[((size_t)hh * (HD*HD) + (size_t)d * 128 + e) * NBLKP + nb] = (bf16_t)acc[t][r];
        }
    }
}

// ---------------- block mixing as WMMA GEMM: out[o,de] = sum_i wblkb[o,i]*kvT[h,de,i] -> kv2T[h,o,e,d] ----------------
// grid.x = 16384/64, grid.y = 3 (10 o-tiles over 12 wave slots), grid.z = NH, block = 128
__global__ __launch_bounds__(128) void mix_kernel(const bf16_t* __restrict__ wblkb,
                                                  const bf16_t* __restrict__ kvT,
                                                  bf16_t* __restrict__ kv2T) {
    const int lane = threadIdx.x & 31;
    const int wave = threadIdx.x >> 5;
    const int mtile = blockIdx.y * 4 + wave;
    if (mtile * 16 >= NBLKP) return;           // wave-uniform
    const int hh = blockIdx.z;
    const int m0 = mtile * 16;                 // o base
    const int n0 = blockIdx.x * 64;            // de base
    const int col = lane & 15;
    const int half = lane >> 4;
    v8f acc[4] = {};
    const bf16_t* arow = wblkb + (size_t)(m0 + col) * NBLKP;
    const bf16_t* Bb = kvT + (size_t)hh * (HD*HD) * NBLKP;
#pragma unroll
    for (int k0 = 0; k0 < NBLKP; k0 += 32) {
        v16bf a = cat8(*(const v8bf*)(arow + k0 + 8*half),
                       *(const v8bf*)(arow + k0 + 16 + 8*half));
#pragma unroll
        for (int t = 0; t < 4; ++t) {
            const bf16_t* br = Bb + (size_t)(n0 + t*16 + col) * NBLKP + k0;
            v16bf b = cat8(*(const v8bf*)(br + 8*half), *(const v8bf*)(br + 16 + 8*half));
            acc[t] = wmma_bf16(a, b, acc[t]);
        }
    }
#pragma unroll
    for (int t = 0; t < 4; ++t) {
        const int de = n0 + t*16 + col;
        const int d = de >> 7;
        const int e = de & 127;
#pragma unroll
        for (int r = 0; r < 8; ++r) {
            const int o = m0 + r + 8*half;
            if (o < NBLK)
                kv2T[(((size_t)hh * NBLK + o) * 128 + e) * 128 + d] = (bf16_t)acc[t][r];
        }
    }
}

// ---------------- qk[h,nb,p] = dot(q_norm[token, h*HD:..], ksum[h,nb,:]) ----------------
__global__ void qk_kernel(const float* __restrict__ q,
                          const float* __restrict__ ksum,
                          float* __restrict__ qk) {
    int idx = blockIdx.x * 256 + threadIdx.x;
    if (idx >= NH * NBLK * PTOK) return;
    const int p = idx % PTOK;
    const int nb = (idx / PTOK) % NBLK;
    const int hh = idx / (PTOK * NBLK);
    const int fb = nb / (HB*WB), rn = nb % (HB*WB), hb = rn / WB, wb = rn % WB;
    const int p1 = p / (P2C*P3C), rp = p % (P2C*P3C), p2 = rp / P3C, p3 = rp % P3C;
    const int n = (fb*P1C + p1) * (H_T*W_T) + (hb*P2C + p2) * W_T + (wb*P3C + p3);
    const float* qr = q + (size_t)n * DIM + hh * HD;
    const float* ks = ksum + ((size_t)hh * NBLK + nb) * HD;
    float s = 0.f;
    for (int d = 0; d < HD; ++d) s += qr[d] * ks[d];
    qk[idx] = s;
}

// ---------------- norm[h,o,p] = sum_i wblk[o,i]*qk[h,i,p] + EPS ----------------
__global__ void norm_kernel(const float* __restrict__ wblk,
                            const float* __restrict__ qk,
                            float* __restrict__ normb) {
    int idx = blockIdx.x * 256 + threadIdx.x;
    if (idx >= NH * NBLK * PTOK) return;
    const int p = idx % PTOK;
    const int o = (idx / PTOK) % NBLK;
    const int hh = idx / (PTOK * NBLK);
    const float* wr = wblk + o * NBLK;
    float s = 0.f;
    for (int i = 0; i < NBLK; ++i) s += wr[i] * qk[((size_t)hh * NBLK + i) * PTOK + p];
    normb[idx] = s + EPSF;
}

// ---------------- out[p,e] = (qrb[p,:] . kv2[:,e]) / norm[p] -> attn bf16 token layout ----------------
__global__ __launch_bounds__(256) void out_kernel(const bf16_t* __restrict__ qrb,
                                                  const bf16_t* __restrict__ kv2T,
                                                  const float* __restrict__ normb,
                                                  bf16_t* __restrict__ attn) {
    const size_t blk = blockIdx.x;
    const int hh = (int)(blk / NBLK);
    const int nb = (int)(blk % NBLK);
    const int lane = threadIdx.x & 31;
    const int wave = threadIdx.x >> 5;
    const int col = lane & 15;
    const int half = lane >> 4;
    const bf16_t* Ab = qrb + blk * 16384;     // [p][d]
    const bf16_t* Bb = kv2T + blk * 16384;    // [e][d]
    const int m0 = wave * 16;
    v8f acc[8] = {};
    const bf16_t* arow = Ab + (size_t)(m0 + col) * 128;
#pragma unroll
    for (int k0 = 0; k0 < 128; k0 += 32) {
        v16bf a = cat8(*(const v8bf*)(arow + k0 + 8*half),
                       *(const v8bf*)(arow + k0 + 16 + 8*half));
#pragma unroll
        for (int t = 0; t < 8; ++t) {
            const bf16_t* br = Bb + (size_t)(t*16 + col) * 128 + k0;
            v16bf b = cat8(*(const v8bf*)(br + 8*half), *(const v8bf*)(br + 16 + 8*half));
            acc[t] = wmma_bf16(a, b, acc[t]);
        }
    }
    const int fb = nb / (HB*WB), rn = nb % (HB*WB), hb = rn / WB, wb = rn % WB;
#pragma unroll
    for (int t = 0; t < 8; ++t) {
        const int e = t*16 + col;
#pragma unroll
        for (int r = 0; r < 8; ++r) {
            const int p = m0 + r + 8*half;
            if (p < PTOK) {
                const float nv = normb[((size_t)hh * NBLK + nb) * PTOK + p];
                const int p1 = p / (P2C*P3C), rp = p % (P2C*P3C), p2 = rp / P3C, p3 = rp % P3C;
                const int n = (fb*P1C + p1) * (H_T*W_T) + (hb*P2C + p2) * W_T + (wb*P3C + p3);
                attn[(size_t)n * DIM + hh * HD + e] = (bf16_t)(acc[t][r] / nv);
            }
        }
    }
}

// ---------------- host-side launch ----------------
extern "C" void kernel_launch(void* const* d_in, const int* in_sizes, int n_in,
                              void* d_out, int out_size, void* d_ws, size_t ws_size,
                              hipStream_t stream) {
    const float* x    = (const float*)d_in[0];
    const float* fcos = (const float*)d_in[3];
    const float* fsin = (const float*)d_in[4];
    const float* wq   = (const float*)d_in[5];
    const float* bq   = (const float*)d_in[6];
    const float* wk   = (const float*)d_in[7];
    const float* bk   = (const float*)d_in[8];
    const float* wv   = (const float*)d_in[9];
    const float* bv   = (const float*)d_in[10];
    const float* wo   = (const float*)d_in[11];
    const float* bo   = (const float*)d_in[12];
    const float* nqw  = (const float*)d_in[13];
    const float* nkw  = (const float*)d_in[14];
    float* outp = (float*)d_out;

    char* ws = (char*)d_ws;
    size_t off = 0;
    auto alloc = [&](size_t bytes) -> void* {
        void* p = ws + off;
        off += (bytes + 255) & ~(size_t)255;
        return p;
    };
    const size_t ND   = (size_t)NTOK * DIM;
    const size_t NDP  = (size_t)MPAD * DIM;
    const size_t WSZ  = (size_t)DIM * DIM;
    const size_t BLK  = (size_t)NH * NBLK * 128 * 128;
    const size_t KVT  = (size_t)NH * HD * HD * NBLKP;

    bf16_t* xb    = (bf16_t*)alloc(NDP * 2);
    bf16_t* wqb   = (bf16_t*)alloc(WSZ * 2);
    bf16_t* wkb   = (bf16_t*)alloc(WSZ * 2);
    bf16_t* wvb   = (bf16_t*)alloc(WSZ * 2);
    bf16_t* wob   = (bf16_t*)alloc(WSZ * 2);
    float*  qf    = (float*)alloc(ND * 4);
    float*  kf    = (float*)alloc(ND * 4);
    float*  vf    = (float*)alloc(ND * 4);
    bf16_t* qrb   = (bf16_t*)alloc(BLK * 2);
    bf16_t* krbT  = (bf16_t*)alloc(BLK * 2);
    bf16_t* vbT   = (bf16_t*)alloc(BLK * 2);
    float*  ksum  = (float*)alloc((size_t)NH * NBLK * HD * 4);
    bf16_t* kvT   = (bf16_t*)alloc(KVT * 2);
    bf16_t* kv2T  = (bf16_t*)alloc(BLK * 2);
    float*  wblk  = (float*)alloc((size_t)NBLK * NBLK * 4);
    bf16_t* wblkb = (bf16_t*)alloc((size_t)NBLKP * NBLKP * 2);
    float*  qkb   = (float*)alloc((size_t)NH * NBLK * PTOK * 4);
    float*  nrm   = (float*)alloc((size_t)NH * NBLK * PTOK * 4);
    bf16_t* attn  = (bf16_t*)alloc(NDP * 2);

    auto gblocks = [](size_t n) { return (unsigned)((n + 255) / 256); };

    // zero padded regions (pad rows/cols must contribute 0 to WMMA)
    zero_u32<<<gblocks(BLK / 2), 256, 0, stream>>>((unsigned int*)qrb,  BLK / 2);
    zero_u32<<<gblocks(BLK / 2), 256, 0, stream>>>((unsigned int*)krbT, BLK / 2);
    zero_u32<<<gblocks(BLK / 2), 256, 0, stream>>>((unsigned int*)vbT,  BLK / 2);
    zero_u32<<<gblocks(KVT / 2), 256, 0, stream>>>((unsigned int*)kvT,  KVT / 2);
    zero_u32<<<gblocks((size_t)NH * NBLK * HD), 256, 0, stream>>>((unsigned int*)ksum, (size_t)NH * NBLK * HD);
    zero_u32<<<gblocks((NDP - ND) / 2), 256, 0, stream>>>((unsigned int*)(xb + ND),   (NDP - ND) / 2);
    zero_u32<<<gblocks((NDP - ND) / 2), 256, 0, stream>>>((unsigned int*)(attn + ND), (NDP - ND) / 2);

    // bf16 conversions
    cvt_f32_bf16<<<gblocks(ND), 256, 0, stream>>>(x,  xb,  ND);
    cvt_f32_bf16<<<gblocks(WSZ), 256, 0, stream>>>(wq, wqb, WSZ);
    cvt_f32_bf16<<<gblocks(WSZ), 256, 0, stream>>>(wk, wkb, WSZ);
    cvt_f32_bf16<<<gblocks(WSZ), 256, 0, stream>>>(wv, wvb, WSZ);
    cvt_f32_bf16<<<gblocks(WSZ), 256, 0, stream>>>(wo, wob, WSZ);

    // W_BLK on device (+ padded bf16 copy)
    wblk_kernel<<<NBLK, 256, 0, stream>>>(wblk);
    wblk_to_bf16<<<gblocks((size_t)NBLKP * NBLKP), 256, 0, stream>>>(wblk, wblkb);

    // QKV projections (WMMA, 64x64 per wave)
    dim3 ggrid(DIM / 64, (MPAD + 255) / 256);
    gemm_bf16<<<ggrid, 128, 0, stream>>>(xb, wqb, bq, qf, MPAD, NTOK, DIM, DIM);
    gemm_bf16<<<ggrid, 128, 0, stream>>>(xb, wkb, bk, kf, MPAD, NTOK, DIM, DIM);
    gemm_bf16<<<ggrid, 128, 0, stream>>>(xb, wvb, bv, vf, MPAD, NTOK, DIM, DIM);

    // fused norm/relu/rope/block-scatter/ksum
    post_kernel<<<NTOK, 256, 0, stream>>>(qf, kf, vf, nqw, nkw, fcos, fsin,
                                          qrb, krbT, vbT, ksum);

    // per-block K^T V (WMMA) -> K-major kvT
    kv_kernel<<<NH * NBLK, 256, 0, stream>>>(krbT, vbT, kvT);

    // block mixing as WMMA GEMM
    dim3 mgrid((HD * HD) / 64, (NBLKP / 16 + 3) / 4, NH);
    mix_kernel<<<mgrid, 128, 0, stream>>>(wblkb, kvT, kv2T);

    // normalizer path
    qk_kernel<<<gblocks((size_t)NH * NBLK * PTOK), 256, 0, stream>>>(qf, ksum, qkb);
    norm_kernel<<<gblocks((size_t)NH * NBLK * PTOK), 256, 0, stream>>>(wblk, qkb, nrm);

    // per-block Q_rope * kv / norm (WMMA) -> attn (token layout, bf16)
    out_kernel<<<NH * NBLK, 256, 0, stream>>>(qrb, kv2T, nrm, attn);

    // final output projection (WMMA) -> f32 d_out
    gemm_bf16<<<ggrid, 128, 0, stream>>>(attn, wob, bo, outp, MPAD, NTOK, DIM, DIM);
}
